// Quantizing_cossim_41635412968057
// MI455X (gfx1250) — matrix operand order repaired and checked by reading, not compiled, not run on hardware
//
#include <hip/hip_runtime.h>
#include <hip/hip_bf16.h>

typedef __bf16 v16bf __attribute__((ext_vector_type(16)));
typedef float  v8f   __attribute__((ext_vector_type(8)));
typedef float  v4f   __attribute__((ext_vector_type(4)));

#define E_DIM  512
#define KC_CNT 16   // E_DIM / 32

// ---------------- kernel 1: 1/||w_q|| per row (one wave per row) ----------------
__global__ __launch_bounds__(256) void wnorm_kernel(const float* __restrict__ w,
                                                    float* __restrict__ wInv, int Q) {
  int wave = threadIdx.x >> 5, lane = threadIdx.x & 31;
  int row = blockIdx.x * 8 + wave;
  if (row >= Q) return;
  const float* p = w + (size_t)row * E_DIM;
  float s = 0.f;
#pragma unroll
  for (int k = 0; k < E_DIM / 32; ++k) { float v = p[lane + 32 * k]; s = fmaf(v, v, s); }
#pragma unroll
  for (int o = 16; o > 0; o >>= 1) s += __shfl_xor(s, o, 32);
  if (lane == 0) wInv[row] = rsqrtf(fmaxf(s, 1e-24f));
}

// ---------------- kernel 2: pack normalized w into B-fragment hi/lo bf16 --------
// B (32x16, KxN) fragment layout: lane l holds column N = l&15,
// half-slot h (0..15) holds K = h + 16*(l>>4).
__global__ __launch_bounds__(256) void wpack_kernel(const float* __restrict__ w,
                                                    const float* __restrict__ wInv,
                                                    __bf16* __restrict__ bhi,
                                                    __bf16* __restrict__ blo, int Q) {
  int t    = blockIdx.x * 256 + threadIdx.x;
  int lane = t & 31;
  int fb   = t >> 5;          // fragment block = qt*KC_CNT + kc
  int qt   = fb >> 4;
  int kc   = fb & 15;
  int q    = qt * 16 + (lane & 15);
  if (q >= Q) return;
  int k0 = kc * 32 + ((lane >> 4) << 4);
  const float* p = w + (size_t)q * E_DIM + k0;
  float inv = wInv[q];
  v16bf hv, lv;
#pragma unroll
  for (int h = 0; h < 16; ++h) {
    float v = p[h] * inv;
    __bf16 hb = (__bf16)v;
    hv[h] = hb;
    lv[h] = (__bf16)(v - (float)hb);
  }
  size_t off = (size_t)fb * 512 + (size_t)lane * 16;
  *(v16bf*)(bhi + off) = hv;
  *(v16bf*)(blo + off) = lv;
}

// ---------------- kernel 3: streaming GEMM + running argmax + gather ------------
__global__ __launch_bounds__(256) void qmain_kernel(const float* __restrict__ x,
                                                    const float* __restrict__ w,
                                                    const __bf16* __restrict__ bhi,
                                                    const __bf16* __restrict__ blo,
                                                    float* __restrict__ outData,
                                                    float* __restrict__ outIdx, int Q) {
  __shared__ int sIdx[128];
  const int wave = threadIdx.x >> 5, lane = threadIdx.x & 31;
  const int m = lane & 15, g = lane >> 4;
  const int rowBase = (blockIdx.x * 8 + wave) * 16;

  // A (16x32, MxK) fragment layout: lane l holds row M = l&15,
  // half h holds K = h + 16*(l>>4).
  v16bf aHi[KC_CNT], aLo[KC_CNT];
  {
    const float* xp = x + (size_t)(rowBase + m) * E_DIM;
#pragma unroll
    for (int kc = 0; kc < KC_CNT; ++kc) {
      const float* p = xp + kc * 32 + 16 * g;
      v4f f0 = *(const v4f*)(p);
      v4f f1 = *(const v4f*)(p + 4);
      v4f f2 = *(const v4f*)(p + 8);
      v4f f3 = *(const v4f*)(p + 12);
#pragma unroll
      for (int e = 0; e < 4; ++e) {
        float v0 = f0[e], v1 = f1[e], v2 = f2[e], v3 = f3[e];
        __bf16 h0 = (__bf16)v0, h1 = (__bf16)v1, h2 = (__bf16)v2, h3 = (__bf16)v3;
        aHi[kc][e]      = h0; aLo[kc][e]      = (__bf16)(v0 - (float)h0);
        aHi[kc][4 + e]  = h1; aLo[kc][4 + e]  = (__bf16)(v1 - (float)h1);
        aHi[kc][8 + e]  = h2; aLo[kc][8 + e]  = (__bf16)(v2 - (float)h2);
        aHi[kc][12 + e] = h3; aLo[kc][12 + e] = (__bf16)(v3 - (float)h3);
      }
    }
  }

  float bestV[8];
  int   bestI[8];
#pragma unroll
  for (int j = 0; j < 8; ++j) { bestV[j] = -3.4e38f; bestI[j] = 0; }

  const int nQT = Q >> 4;
  for (int qt = 0; qt < nQT; ++qt) {
    v8f c = {};
    const __bf16* ph = bhi + ((size_t)qt * 16) * 512 + (size_t)lane * 16;
    const __bf16* pl = blo + ((size_t)qt * 16) * 512 + (size_t)lane * 16;
#pragma unroll
    for (int kc = 0; kc < KC_CNT; ++kc) {
      v16bf bH = *(const v16bf*)(ph + (size_t)kc * 512);
      v16bf bL = *(const v16bf*)(pl + (size_t)kc * 512);
      // split-fp32 emulation: hi*hi + hi*lo + lo*hi, f32 accumulate
      c = __builtin_amdgcn_wmma_f32_16x16x32_bf16(false, aHi[kc], false, bH, (short)0, c, false, false);
      c = __builtin_amdgcn_wmma_f32_16x16x32_bf16(false, aHi[kc], false, bL, (short)0, c, false, false);
      c = __builtin_amdgcn_wmma_f32_16x16x32_bf16(false, aLo[kc], false, bH, (short)0, c, false, false);
    }
    int col = (qt << 4) + m;
#pragma unroll
    for (int j = 0; j < 8; ++j)
      if (c[j] > bestV[j]) { bestV[j] = c[j]; bestI[j] = col; }
  }

  // argmax across the 16 columns (lanes of each 16-lane half); ties -> smaller q
#pragma unroll
  for (int j = 0; j < 8; ++j) {
#pragma unroll
    for (int s = 1; s < 16; s <<= 1) {
      float ov = __shfl_xor(bestV[j], s, 32);
      int   oi = __shfl_xor(bestI[j], s, 32);
      if (ov > bestV[j] || (ov == bestV[j] && oi < bestI[j])) { bestV[j] = ov; bestI[j] = oi; }
    }
    if (m == 0) {
      int r = j + 8 * g;                 // row within this wave's 16-row tile
      sIdx[wave * 16 + r] = bestI[j];
      outIdx[rowBase + r] = (float)bestI[j];
    }
  }
  __syncthreads();

  // gather q_data = weight[q_idx] for this block's 128 rows (coalesced float4)
  const int blockRow0 = blockIdx.x * 128;
  for (int i = threadIdx.x; i < 128 * (E_DIM / 4); i += 256) {
    int r  = i >> 7;                      // E_DIM/4 == 128
    int k4 = i & 127;
    int qi = sIdx[r];
    v4f v = ((const v4f*)(w + (size_t)qi * E_DIM))[k4];
    ((v4f*)(outData + (size_t)(blockRow0 + r) * E_DIM))[k4] = v;
  }
}

extern "C" void kernel_launch(void* const* d_in, const int* in_sizes, int n_in,
                              void* d_out, int out_size, void* d_ws, size_t ws_size,
                              hipStream_t stream) {
  const float* x = (const float*)d_in[0];
  const float* w = (const float*)d_in[1];
  const int B = in_sizes[0] / E_DIM;
  const int Q = in_sizes[1] / E_DIM;

  float* outData = (float*)d_out;
  float* outIdx  = outData + (size_t)B * E_DIM;

  char*  ws   = (char*)d_ws;
  float* wInv = (float*)ws;
  size_t o1   = (((size_t)Q * sizeof(float)) + 255) & ~(size_t)255;
  __bf16* bhi = (__bf16*)(ws + o1);
  __bf16* blo = bhi + (size_t)Q * E_DIM;

  wnorm_kernel<<<(Q + 7) / 8, 256, 0, stream>>>(w, wInv, Q);
  const int fbCount = (Q / 16) * KC_CNT;
  wpack_kernel<<<(fbCount * 32 + 255) / 256, 256, 0, stream>>>(w, wInv, bhi, blo, Q);
  qmain_kernel<<<B / 128, 256, 0, stream>>>(x, w, bhi, blo, outData, outIdx, Q);
}
